// Retrieval2D_68667937128504
// MI455X (gfx1250) — matrix operand order repaired and checked by reading, not compiled, not run on hardware
//
#include <hip/hip_runtime.h>
#include <hip/hip_bf16.h>

typedef __attribute__((ext_vector_type(16))) __bf16 v16bf;
typedef __attribute__((ext_vector_type(8)))  __bf16 v8bf;
typedef __attribute__((ext_vector_type(8)))  float  v8f;
typedef __attribute__((ext_vector_type(4)))  float  f32x4;
typedef __attribute__((ext_vector_type(4)))  int    v4i;

#define BQ 256       // queries (GEMM M)
#define NB 50000     // bank rows (GEMM N)
#define DD 2048      // depth (GEMM K)
#define NTILE 64     // N columns per workgroup
#define KSTEP 64     // K per staging step (2 x K=32 WMMA slices)
#define KITERS (DD / KSTEP)

#define GLOBAL_AS __attribute__((address_space(1)))
#define LDS_AS    __attribute__((address_space(3)))

// Async global->LDS path (gfx1250 ASYNCcnt machinery), with a guarded fallback
// so the TU compiles even if the toolchain lacks the builtins.
#if defined(__HIP_DEVICE_COMPILE__) &&                                   \
    __has_builtin(__builtin_amdgcn_global_load_async_to_lds_b128) &&     \
    __has_builtin(__builtin_amdgcn_s_wait_asynccnt)
#define HAVE_ASYNC_LDS 1
#else
#define HAVE_ASYNC_LDS 0
#endif

__device__ __forceinline__ void async_wait_all() {
#if HAVE_ASYNC_LDS
    __builtin_amdgcn_s_wait_asynccnt(0);
#endif
}

// 16-byte global(bf16) -> LDS copy
__device__ __forceinline__ void copy16B_g2lds(const __bf16* g, __bf16* l) {
#if HAVE_ASYNC_LDS
    __builtin_amdgcn_global_load_async_to_lds_b128(
        (GLOBAL_AS v4i*)(unsigned long long)(uintptr_t)g,
        (LDS_AS v4i*)(unsigned)(uintptr_t)l,
        0, 0);
#else
    *(v8bf*)l = *(const v8bf*)g;
#endif
}

// ---------------------------------------------------------------------------
// Prep: query norms + fp32 -> bf16 conversion of queries into workspace.
// ---------------------------------------------------------------------------
__global__ __launch_bounds__(256) void prep_kernel(const float* __restrict__ q,
                                                   float* __restrict__ qn,
                                                   __hip_bfloat16* __restrict__ qbf) {
    __shared__ float red[256];
    const int b = blockIdx.x;
    const int t = threadIdx.x;
    const float* row = q + (size_t)b * DD;
    float s = 0.f;
#pragma unroll
    for (int i = 0; i < DD / 256; ++i) {
        const int idx = t + i * 256;   // coalesced
        const float v = row[idx];
        s += v * v;
        qbf[(size_t)b * DD + idx] = __float2bfloat16(v);
    }
    red[t] = s;
    __syncthreads();
    for (int off = 128; off > 0; off >>= 1) {
        if (t < off) red[t] += red[t + off];
        __syncthreads();
    }
    if (t == 0) qn[b] = sqrtf(red[0]);
}

// ---------------------------------------------------------------------------
// Fused GEMM: scores[m,n] = (Q . B^T) / max(qn*bn, eps).
// One WG = all 256 M rows x 64 N cols so the bank streams from HBM once.
// Double-buffered LDS; A tiles staged with async global->LDS loads; bank
// row norms computed on the fly during fp32->bf16 staging of B.
// ---------------------------------------------------------------------------
__global__ __launch_bounds__(256) void gemm_kernel(const float* __restrict__ bank,
                                                   const __hip_bfloat16* __restrict__ qbf_,
                                                   const float* __restrict__ qn,
                                                   float* __restrict__ scores) {
    __shared__ __align__(16) __bf16 Alds[2][BQ * KSTEP];     // 2 x 32 KB, [m][k]
    __shared__ __align__(16) __bf16 Blds[2][NTILE * KSTEP];  // 2 x  8 KB, [n][k]
    __shared__ float bn_lds[NTILE];
    __shared__ float qn_lds[BQ];

    const __bf16* __restrict__ qbf = (const __bf16*)qbf_;

    const int t    = threadIdx.x;
    const int wave = t >> 5;
    const int lane = t & 31;
    const int n0   = blockIdx.x * NTILE;

    qn_lds[t] = qn[t];   // one element per thread; barriers below cover it

    // wave tiling: wm in 0..3 (64 M rows each), wn in 0..1 (32 N cols each)
    const int  wm  = wave & 3;
    const int  wn  = wave >> 2;
    const int  l16 = lane & 15;
    const bool hiL = (lane >= 16);

    // B staging: thread -> bank row nb = t/4, 16 contiguous floats at (t%4)*16
    const int  nb      = t >> 2;
    const int  kbth    = (t & 3) << 4;
    const int  n_glb   = n0 + nb;
    const bool brow_ok = (n_glb < NB);
    const float* bank_row = bank + (size_t)n_glb * DD + kbth;

    v8f   acc[4][2] = {};
    float bsq = 0.f;
    float fv[16];

    // ---- pipeline prologue: stage K-step 0 into buffer 0 ----
#pragma unroll
    for (int i = 0; i < 8; ++i) {
        const int c = i * 256 + t;           // 2048 16B-chunks of the A tile
        const int m = c >> 3;
        const int k = (c & 7) << 3;
        copy16B_g2lds(qbf + (size_t)m * DD + k, &Alds[0][m * KSTEP + k]);
    }
    {
#pragma unroll
        for (int i = 0; i < 16; ++i) fv[i] = 0.f;
        if (brow_ok) {
            const f32x4* src = (const f32x4*)bank_row;
#pragma unroll
            for (int i = 0; i < 4; ++i) {
                f32x4 f = __builtin_nontemporal_load(src + i);
                fv[4 * i + 0] = f[0]; fv[4 * i + 1] = f[1];
                fv[4 * i + 2] = f[2]; fv[4 * i + 3] = f[3];
            }
        }
#pragma unroll
        for (int i = 0; i < 16; ++i) bsq += fv[i] * fv[i];
        v8bf h0, h1;
#pragma unroll
        for (int i = 0; i < 8; ++i) { h0[i] = (__bf16)fv[i]; h1[i] = (__bf16)fv[8 + i]; }
        *(v8bf*)(&Blds[0][nb * KSTEP + kbth]) = h0;
        *(v8bf*)(&Blds[0][nb * KSTEP + kbth + 8]) = h1;
    }
    async_wait_all();
    __syncthreads();

    // ---- main pipelined K loop ----
    for (int ki = 0; ki < KITERS; ++ki) {
        const int  kk       = ki * KSTEP;
        const int  buf      = ki & 1;
        const int  nxt      = buf ^ 1;
        const bool has_next = (ki + 1) < KITERS;

        // prefetch next K-step: async A -> LDS[nxt], B fp32 -> registers
        if (has_next) {
#pragma unroll
            for (int i = 0; i < 8; ++i) {
                const int c = i * 256 + t;
                const int m = c >> 3;
                const int k = (c & 7) << 3;
                copy16B_g2lds(qbf + (size_t)m * DD + (kk + KSTEP) + k,
                              &Alds[nxt][m * KSTEP + k]);
            }
#pragma unroll
            for (int i = 0; i < 16; ++i) fv[i] = 0.f;
            if (brow_ok) {
                const f32x4* src = (const f32x4*)(bank_row + (kk + KSTEP));
#pragma unroll
                for (int i = 0; i < 4; ++i) {
                    f32x4 f = __builtin_nontemporal_load(src + i);
                    fv[4 * i + 0] = f[0]; fv[4 * i + 1] = f[1];
                    fv[4 * i + 2] = f[2]; fv[4 * i + 3] = f[3];
                }
            }
        }

        // ---- compute current buffer: 2 K-slices x (4 M x 2 N) WMMAs ----
        const __bf16* Ab = Alds[buf];
        const __bf16* Bb = Blds[buf];
#pragma unroll
        for (int ks = 0; ks < 2; ++ks) {
            const int kb32 = ks * 32;
            v16bf afrag[4];
#pragma unroll
            for (int tm = 0; tm < 4; ++tm) {
                // A 16-bit layout: lanes 0-15 -> K {0..7,16..23}, lanes 16-31 -> K {8..15,24..31}
                const int m = wm * 64 + tm * 16 + l16;
                const __bf16* p = Ab + m * KSTEP + kb32 + (hiL ? 8 : 0);
                v8bf lo = *(const v8bf*)p;
                v8bf hi = *(const v8bf*)(p + 16);
                afrag[tm] = __builtin_shufflevector(lo, hi,
                    0, 1, 2, 3, 4, 5, 6, 7, 8, 9, 10, 11, 12, 13, 14, 15);
            }
#pragma unroll
            for (int tn = 0; tn < 2; ++tn) {
                // B layout: lane%16 = column, lanes 0-15 hold K {0..7,16..23}, lanes 16-31 hold K {8..15,24..31}
                const int n = wn * 32 + tn * 16 + l16;
                const __bf16* p = Bb + n * KSTEP + kb32 + (hiL ? 8 : 0);
                v8bf lo = *(const v8bf*)p;
                v8bf hi = *(const v8bf*)(p + 16);
                v16bf bfrag = __builtin_shufflevector(lo, hi,
                    0, 1, 2, 3, 4, 5, 6, 7, 8, 9, 10, 11, 12, 13, 14, 15);
#pragma unroll
                for (int tm = 0; tm < 4; ++tm) {
                    acc[tm][tn] = __builtin_amdgcn_wmma_f32_16x16x32_bf16(
                        false, afrag[tm], false, bfrag,
                        (short)0, acc[tm][tn], false, false);
                }
            }
        }

        // convert + store next B tile (after compute so loads are covered)
        if (has_next) {
#pragma unroll
            for (int i = 0; i < 16; ++i) bsq += fv[i] * fv[i];
            v8bf h0, h1;
#pragma unroll
            for (int i = 0; i < 8; ++i) { h0[i] = (__bf16)fv[i]; h1[i] = (__bf16)fv[8 + i]; }
            *(v8bf*)(&Blds[nxt][nb * KSTEP + kbth]) = h0;
            *(v8bf*)(&Blds[nxt][nb * KSTEP + kbth + 8]) = h1;
        }

        async_wait_all();
        __syncthreads();
    }

    // ---- bank row norms: 4 staging threads per row hold partial sums ----
    bsq += __shfl_xor(bsq, 1);
    bsq += __shfl_xor(bsq, 2);
    if ((t & 3) == 0) bn_lds[nb] = sqrtf(bsq);
    __syncthreads();

    // ---- epilogue: scale by 1/max(qn*bn, eps) and store ----
#pragma unroll
    for (int tn = 0; tn < 2; ++tn) {
        const int nl = wn * 32 + tn * 16 + l16;
        const int n  = n0 + nl;
        if (n >= NB) continue;
        const float bn = bn_lds[nl];
#pragma unroll
        for (int tm = 0; tm < 4; ++tm) {
#pragma unroll
            for (int r = 0; r < 8; ++r) {
                const int m = wm * 64 + tm * 16 + r + (hiL ? 8 : 0);
                const float denom = fmaxf(qn_lds[m] * bn, 1e-12f);
                scores[(size_t)m * NB + n] = acc[tm][tn][r] / denom;
            }
        }
    }
}

// ---------------------------------------------------------------------------
// Argmax per query row (first-max semantics on ties).
// ---------------------------------------------------------------------------
__global__ __launch_bounds__(256) void argmax_kernel(const float* __restrict__ scores,
                                                     float* __restrict__ out_idx) {
    __shared__ float vred[256];
    __shared__ int   ired[256];
    const int b = blockIdx.x;
    const int t = threadIdx.x;
    const float* row = scores + (size_t)b * NB;
    float best = -3.402823466e38f;
    int   bi   = 0;
    for (int i = t; i < NB; i += 256) {
        const float v = row[i];
        if (v > best) { best = v; bi = i; }
    }
    vred[t] = best; ired[t] = bi;
    __syncthreads();
    for (int off = 128; off > 0; off >>= 1) {
        if (t < off) {
            const float v2 = vred[t + off];
            const int   i2 = ired[t + off];
            if (v2 > vred[t] || (v2 == vred[t] && i2 < ired[t])) {
                vred[t] = v2; ired[t] = i2;
            }
        }
        __syncthreads();
    }
    if (t == 0) out_idx[b] = (float)ired[0];
}

// ---------------------------------------------------------------------------
// Launch: prep (norms + bf16 queries) -> fused GEMM -> argmax.
// ws layout: [0,1KB) q_norm f32[256]; [1KB, 1KB+1MB) queries bf16[256*2048].
// d_out: [0,256) best_idx as f32; [256, ...) scores f32[256*50000].
// ---------------------------------------------------------------------------
extern "C" void kernel_launch(void* const* d_in, const int* in_sizes, int n_in,
                              void* d_out, int out_size, void* d_ws, size_t ws_size,
                              hipStream_t stream) {
    const float* queries = (const float*)d_in[0];
    const float* bank    = (const float*)d_in[1];

    float*          out    = (float*)d_out;
    float*          qn     = (float*)d_ws;
    __hip_bfloat16* qbf    = (__hip_bfloat16*)((char*)d_ws + 1024);
    float*          scores = out + BQ;

    prep_kernel<<<BQ, 256, 0, stream>>>(queries, qn, qbf);
    gemm_kernel<<<(NB + NTILE - 1) / NTILE, 256, 0, stream>>>(bank, qbf, qn, scores);
    argmax_kernel<<<BQ, 256, 0, stream>>>(scores, out);
}
